// SimpleRNN_87591563035128
// MI455X (gfx1250) — matrix operand order repaired and checked by reading, not compiled, not run on hardware
//
#include <hip/hip_runtime.h>

#define V_DIM 4096
#define H_DIM 256
#define O_DIM 512
#define B_DIM 64
#define T_DIM 512
#define LDSS  264   // padded LDS row stride (bf16 elems): 528B/row = 4-bank rotation, b128-friendly

typedef __bf16 bf16_t;
typedef bf16_t v16bf __attribute__((ext_vector_type(16)));
typedef bf16_t v8bf  __attribute__((ext_vector_type(8)));
typedef float  v8f   __attribute__((ext_vector_type(8)));
typedef float  v4f   __attribute__((ext_vector_type(4)));
typedef float  v2f   __attribute__((ext_vector_type(2)));

// tanh(x) = 1 - 2/(exp2(x*2*log2e) + 1); saturates correctly at +-1 for |x| large,
// uses only v_mul + v_exp(TRANS) + v_add + v_rcp(TRANS) + v_fma.
__device__ __forceinline__ float fast_tanh(float x) {
    const float e = __builtin_amdgcn_exp2f(x * 2.88539008177792681472f);
    const float r = __builtin_amdgcn_rcpf(e + 1.0f);
    return __builtin_fmaf(-2.0f, r, 1.0f);
}

// ---------------------------------------------------------------------------
// Phase 1: x-projection gather. pre[t][b][h] = W_ih[h][x[b][t]] + b_ih[h] + b_hh[h]
// (biases folded so the recurrence loop is pure GEMM + tanh). W_ih is 4MB ->
// L2-resident; output 33.5MB streamed once.
// ---------------------------------------------------------------------------
__global__ __launch_bounds__(256) void embed_kernel(const int* __restrict__ x,
                                                    const float* __restrict__ W_ih,
                                                    const float* __restrict__ b_ih,
                                                    const float* __restrict__ b_hh,
                                                    float* __restrict__ pre) {
    const int t = blockIdx.x;
    const int b = blockIdx.y;
    const int h = threadIdx.x;
    const int v = x[(size_t)b * T_DIM + t];
    pre[((size_t)t * B_DIM + b) * H_DIM + h] =
        W_ih[(size_t)h * V_DIM + v] + b_ih[h] + b_hh[h];
}

// ---------------------------------------------------------------------------
// Phase 2: sequential recurrence, ONE workgroup (latency-bound chain).
// Transposed GEMM: h_next^T = W_hh * h^T  (M = hidden_out, N = batch).
//   A = W_hh strip: static bf16 fragments in registers (64 VGPR/wave).
//   B = h^T: h stored [batch][hidden] bf16 in LDS -> per-lane two 16B reads.
//   C: lane owns one batch column, 8 consecutive hidden elems -> vectorized
//      x-proj loads, hs stores (b128) and ONE ds_store_b128 per tile.
// 512 threads = 16 waves; wave = M-tile (hidden strip), 4 N-tiles (batch) each.
// ---------------------------------------------------------------------------
__global__ __launch_bounds__(512, 1) void rnn_kernel(const float* __restrict__ pre,
                                                     const float* __restrict__ W_hh,
                                                     float* __restrict__ hs) {
    __shared__ bf16_t hbuf[B_DIM * LDSS];

    const int tid    = threadIdx.x;
    const int wave   = tid >> 5;        // 0..15 = M-tile (hidden_out strip)
    const int lane   = tid & 31;
    const int laneLo = lane & 15;
    const int hiHalf = lane >> 4;       // 0 or 1
    const int mb     = wave * 16;       // hidden_out base for this wave
    const int hcol   = mb + hiHalf * 8; // first of 8 consecutive hidden elems per lane

    // --- Static A fragments: A[m][k] = W_hh[m][k], 8 chunks of K=32, bf16 ---
    // 16-bit A 16x32 layout: lanes 0-15 row M=lane hold K = {0..7,16..23};
    // lanes 16-31 row M=lane-16 hold K = {8..15,24..31}.
    v16bf Af[8];
    {
        const float* srow = W_hh + (size_t)(mb + laneLo) * H_DIM + hiHalf * 8;
#pragma unroll
        for (int kc = 0; kc < 8; ++kc) {
            const float* src = srow + kc * 32;
            v16bf f;
#pragma unroll
            for (int e = 0; e < 8; ++e) {
                f[e]     = (bf16_t)src[e];
                f[8 + e] = (bf16_t)src[16 + e];
            }
            Af[kc] = f;
        }
    }

    // h0 = 0
    for (int i = tid; i < B_DIM * LDSS; i += 512) hbuf[i] = (bf16_t)0.0f;
    __syncthreads();

    for (int t = 0; t < T_DIM; ++t) {
        const float* pt = pre + (size_t)t * (B_DIM * H_DIM);

        // Issue x-projection loads early; consumed after the WMMA chain so the
        // L2 latency hides under 32 WMMAs. Per lane/tile: 8 consecutive floats.
        v4f p[4][2];
#pragma unroll
        for (int nt = 0; nt < 4; ++nt) {
            const v4f* pp = (const v4f*)&pt[(size_t)(nt * 16 + laneLo) * H_DIM + hcol];
            p[nt][0] = pp[0];
            p[nt][1] = pp[1];
        }
        if (t + 1 < T_DIM)
            __builtin_prefetch(pt + (size_t)(B_DIM * H_DIM) + (size_t)laneLo * H_DIM + hcol, 0, 0);

        // --- K=256 in 8 chunks; 4 independent accumulator chains (batch tiles),
        // B fragments double-buffered one WMMA ahead. ---
        v8f c[4];
#pragma unroll
        for (int nt = 0; nt < 4; ++nt)
#pragma unroll
            for (int i = 0; i < 8; ++i) c[nt][i] = 0.0f;

        // B 32x16 layout: lanes 0-15 col N=batch=lane hold K = {0..7,16..23},
        // lanes 16-31 col N=lane-16 hold K = {8..15,24..31} (matches A).
        v8bf blo[2], bhi[2];
        {
            const v8bf* q = (const v8bf*)&hbuf[laneLo * LDSS + hiHalf * 8];
            blo[0] = q[0];
            bhi[0] = q[2];
        }
#pragma unroll
        for (int it = 0; it < 32; ++it) {
            const int kcur = it >> 2;
            const int ncur = it & 3;
            const int nx   = it + 1;
            if (nx < 32) {
                const int kn = nx >> 2, nn = nx & 3;
                const v8bf* q =
                    (const v8bf*)&hbuf[(nn * 16 + laneLo) * LDSS + kn * 32 + hiHalf * 8];
                blo[nx & 1] = q[0];
                bhi[nx & 1] = q[2];
            }
            v16bf b;
#pragma unroll
            for (int e = 0; e < 8; ++e) {
                b[e]     = blo[it & 1][e];
                b[8 + e] = bhi[it & 1][e];
            }
            c[ncur] = __builtin_amdgcn_wmma_f32_16x16x32_bf16(
                          false, Af[kcur], false, b, (short)0, c[ncur], false, false);
        }

        // Epilogue: add x-proj, tanh, stream f32 to hs (global) BEFORE the
        // barrier; only the 4 packed LDS stores sit between barriers.
        v8bf hb[4];
#pragma unroll
        for (int nt = 0; nt < 4; ++nt) {
            const int batch = nt * 16 + laneLo;
            v4f o0, o1;
#pragma unroll
            for (int i = 0; i < 4; ++i) {
                const float v0 = fast_tanh(c[nt][i]     + p[nt][0][i]);
                const float v1 = fast_tanh(c[nt][i + 4] + p[nt][1][i]);
                o0[i] = v0;
                o1[i] = v1;
                hb[nt][i]     = (bf16_t)v0;
                hb[nt][i + 4] = (bf16_t)v1;
            }
            v4f* o = (v4f*)&hs[((size_t)batch * T_DIM + t) * H_DIM + hcol];
            o[0] = o0;
            o[1] = o1;
        }

        __syncthreads();   // all reads of old h finished
#pragma unroll
        for (int nt = 0; nt < 4; ++nt)
            *(v8bf*)&hbuf[(nt * 16 + laneLo) * LDSS + hcol] = hb[nt];
        __syncthreads();   // new h visible for next step
    }
}

// ---------------------------------------------------------------------------
// Phase 3: out = hs @ W_fc^T + b_fc.  [32768 x 256] x [256 x 512].
// Bandwidth-bound (64MB output write; hs + W_fc are L2-resident after phase 2),
// so keep full f32 precision with v_wmma_f32_16x16x4_f32.
// ---------------------------------------------------------------------------
__global__ __launch_bounds__(256) void fc_kernel(const float* __restrict__ hs,
                                                 const float* __restrict__ W_fc,
                                                 const float* __restrict__ b_fc,
                                                 float* __restrict__ out) {
    const int tid    = threadIdx.x;
    const int wave   = tid >> 5;
    const int lane   = tid & 31;
    const int laneLo = lane & 15;
    const int hiHalf = lane >> 4;
    const int nb     = blockIdx.x * 16;
    const int mb     = (blockIdx.y * 8 + wave) * 16;

    v8f c;
    {
        const float bias = b_fc[nb + laneLo];
#pragma unroll
        for (int i = 0; i < 8; ++i) c[i] = bias;
    }

    // A (16x4 f32): lanes 0-15 row M=lane hold K=k,k+1; lanes 16-31 K=k+2,k+3.
    // B (4x16 f32): mirrored for column N.
    const float* arow = hs   + (size_t)(mb + laneLo) * H_DIM + hiHalf * 2;
    const float* brow = W_fc + (size_t)(nb + laneLo) * H_DIM + hiHalf * 2;
#pragma unroll 8
    for (int k = 0; k < H_DIM; k += 4) {
        const v2f a = *(const v2f*)(arow + k);
        const v2f b = *(const v2f*)(brow + k);
        c = __builtin_amdgcn_wmma_f32_16x16x4_f32(false, a, false, b,
                                                  (short)0, c, false, false);
    }

#pragma unroll
    for (int i = 0; i < 8; ++i) {
        const int m = mb + i + 8 * hiHalf;
        out[(size_t)m * O_DIM + nb + laneLo] = c[i];
    }
}

// ---------------------------------------------------------------------------
extern "C" void kernel_launch(void* const* d_in, const int* in_sizes, int n_in,
                              void* d_out, int out_size, void* d_ws, size_t ws_size,
                              hipStream_t stream) {
    const int*   x    = (const int*)d_in[0];
    const float* W_ih = (const float*)d_in[1];
    const float* W_hh = (const float*)d_in[2];
    const float* b_ih = (const float*)d_in[3];
    const float* b_hh = (const float*)d_in[4];
    const float* W_fc = (const float*)d_in[5];
    const float* b_fc = (const float*)d_in[6];
    float* out = (float*)d_out;

    float* pre = (float*)d_ws;                               // [T][B][H] f32
    float* hs  = pre + (size_t)T_DIM * B_DIM * H_DIM;        // [B*T][H]   f32

    embed_kernel<<<dim3(T_DIM, B_DIM), 256, 0, stream>>>(x, W_ih, b_ih, b_hh, pre);
    rnn_kernel<<<1, 512, 0, stream>>>(pre, W_hh, hs);
    fc_kernel<<<dim3(O_DIM / 16, (B_DIM * T_DIM / 16) / 8), 256, 0, stream>>>(hs, W_fc, b_fc, out);
}
